// SelfAttn_19104014532860
// MI455X (gfx1250) — matrix operand order repaired and checked by reading, not compiled, not run on hardware
//
#include <hip/hip_runtime.h>

// MI455X (gfx1250, wave32) self-attention block.
// f16-in / f32-accumulate WMMA (16x16x32); async global->LDS panel staging;
// sum-only softmax (scores bounded, max-shift mathematically redundant);
// non-temporal stores for the 537MB attention stream.
//
// Workspace (f16):
//   Apk : [B][N/16][32][16]  F packed in WMMA A-layout                  (2 MB)
//   Gpk : [B][N/16][32][16]  G packed in WMMA B-layout                  (2 MB)
//   Hcp : [B][N/32 panels][C][32]  Hv K-panels (16KB each, contiguous)  (16 MB)
//   Wpk : [20][8][32][16]    Wcat packed in A-layout                    (160 KB)

typedef __attribute__((ext_vector_type(16))) _Float16 v16h;
typedef __attribute__((ext_vector_type(8)))  _Float16 v8h;
typedef __attribute__((ext_vector_type(8)))  float    v8f;
typedef __attribute__((ext_vector_type(4)))  float    v4f;

#define WMMA_F16(A, B, C) \
  __builtin_amdgcn_wmma_f32_16x16x32_f16(false, (A), false, (B), (short)0, (C), false, false)

static constexpr int BB = 8;     // batch
static constexpr int CC = 256;   // channels
static constexpr int NN = 4096;  // spatial (64*64)

// ---- CDNA5 async global->LDS (ASYNCcnt-tracked) ---------------------------
typedef __attribute__((address_space(3))) char lds_char;

__device__ __forceinline__ unsigned lds_addr(const void* p) {
  return (unsigned)(unsigned long long)(lds_char*)(void*)(const_cast<void*>(p));
}
__device__ __forceinline__ void async_copy_b128(unsigned lds_off, const void* g) {
  asm volatile("global_load_async_to_lds_b128 %0, %1, off"
               :: "v"(lds_off), "v"((unsigned long long)g) : "memory");
}
__device__ __forceinline__ void wait_async0() {
  asm volatile("s_wait_asynccnt 0" ::: "memory");
}

// ---------------------------------------------------------------------------
// Kernel 0: pack concat(Wf,Wg,Wh) (320x256 f32) into WMMA A-layout f16 tiles.
// A 16x32 f16 (this file's contiguous-k packing): element (m,k) -> lane = m + 16*(k>>4), h = k&15
// ---------------------------------------------------------------------------
__global__ __launch_bounds__(256) void pack_w_kernel(
    const float* __restrict__ Wf, const float* __restrict__ Wg,
    const float* __restrict__ Wh, _Float16* __restrict__ Wpk)
{
  int t = blockIdx.x * 256 + threadIdx.x;
  if (t >= 320 * 256) return;
  int r = t >> 8, c = t & 255;
  float v;
  if (r < 32)       v = Wf[r * 256 + c];
  else if (r < 64)  v = Wg[(r - 32) * 256 + c];
  else              v = Wh[(r - 64) * 256 + c];
  int rt = r >> 4, rr = r & 15;
  int ks = c >> 5, kk = c & 31;
  int lp = rr + 16 * (kk >> 4);
  int hp = kk & 15;
  Wpk[(((rt * 8 + ks) * 32) + lp) * 16 + hp] = (_Float16)v;
}

// ---------------------------------------------------------------------------
// Kernel 1: projections Y = [Wf;Wg;Wh] @ x[b] per (b, 16-col tile of N).
// B 32x16 f16 (this file's contiguous-k packing): element (k,n) -> lane = n + 16*(k>>4), h = k&15.
// C/D f32 16x16: element (m,n) -> lane = n + 16*(m>>3), vgpr = m&7.
// ---------------------------------------------------------------------------
__global__ __launch_bounds__(256) void proj_kernel(
    const float* __restrict__ x, const _Float16* __restrict__ Wpk,
    _Float16* __restrict__ Apk, _Float16* __restrict__ Gpk,
    _Float16* __restrict__ Hcp)
{
  const int b  = blockIdx.x >> 8;
  const int nt = blockIdx.x & 255;

  __shared__ _Float16 xT[16][288];   // [n_rel][c], padded rows

  for (int it = 0; it < 16; ++it) {
    int j = threadIdx.x & 15;
    int c = it * 16 + (threadIdx.x >> 4);
    xT[j][c] = (_Float16)__builtin_nontemporal_load(
        &x[((long)b * CC + c) * NN + nt * 16 + j]);
  }
  __syncthreads();

  const int w = threadIdx.x >> 5, lane = threadIdx.x & 31;
  const int ln = lane & 15, lhi = lane >> 4;

  for (int i = 0; i < 3; ++i) {
    int rt = w + 8 * i;            // wave-uniform
    if (rt >= 20) break;
    v8f acc = {};
    for (int ks = 0; ks < 8; ++ks) {
      v16h a  = *(const v16h*)(Wpk + (((rt * 8 + ks) * 32 + lane) << 4));
      v16h bm = *(const v16h*)(&xT[ln][ks * 32 + 16 * lhi]);
      acc = WMMA_F16(a, bm, acc);
    }
    #pragma unroll
    for (int v = 0; v < 8; ++v) {
      _Float16 hv = (_Float16)acc[v];
      if (rt < 2) {                               // F -> scores A-layout
        int k  = rt * 16 + v + 8 * lhi;
        int lp = ln + 16 * (k >> 4);
        int hp = k & 15;
        Apk[(((long)b * 256 + nt) * 32 + lp) * 16 + hp] = hv;
      } else if (rt < 4) {                        // G -> scores B-layout
        int k  = (rt - 2) * 16 + v + 8 * lhi;
        int lp = ln + 16 * (k >> 4);
        int hp = k & 15;
        Gpk[(((long)b * 256 + nt) * 32 + lp) * 16 + hp] = hv;
      } else {                                    // Hv -> K-panels [b][nt/2][c][32]
        int c = (rt - 4) * 16 + v + 8 * lhi;
        Hcp[(((long)b * 128 + (nt >> 1)) * 256 + c) * 32 + (nt & 1) * 16 + ln] = hv;
      }
    }
  }
}

// ---------------------------------------------------------------------------
// Kernel 2: per (b, 64-row block): scores, softmax (sum of exp), attention
// write (transposed, NT), out^T = attn @ Hv^T with async-staged LDS panels.
// 8 waves = (rt row-tile 0..3) x (ch C-half 0..1).
// ---------------------------------------------------------------------------
__global__ __launch_bounds__(256) void attn_out_kernel(
    const _Float16* __restrict__ Apk, const _Float16* __restrict__ Gpk,
    const _Float16* __restrict__ Hcp, const float* __restrict__ x,
    const float* __restrict__ gamma, float* __restrict__ out,
    float* __restrict__ attention)
{
  const int b  = blockIdx.x >> 6;
  const int nb = blockIdx.x & 63;
  const int tid = threadIdx.x;
  const int w = tid >> 5, lane = tid & 31;
  const int ln = lane & 15, lhi = lane >> 4;
  const int rt = w & 3, ch = w >> 2;
  const int n0 = nb * 64 + rt * 16;
  const int ntile = nb * 4 + rt;

  __shared__ _Float16 ldsH[2][256 * 40];   // Hv panel, rows padded 32->40 halfs
  __shared__ _Float16 ldsG[2][1024];       // 2 Gpk tiles per K-step
  __shared__ _Float16 ldsA[8][16 * 40];    // per-wave attn tile (row-major, padded)
  __shared__ float    ldsStats[4][2][16];  // [rt][ch][row] partial sum-of-exp

  const unsigned ldsHoff = lds_addr(&ldsH[0][0]);
  const unsigned ldsGoff = lds_addr(&ldsG[0][0]);

  const v16h a_s = *(const v16h*)(Apk + (((long)b * 256 + ntile) * 32 + lane) * 16);
  const _Float16* gbase = Gpk + (long)b * 131072 + lane * 16;   // 256 tiles * 512
  const _Float16* hpan  = Hcp + (long)b * 128 * 8192;           // panels, 8192 halfs

  // ---- pass 1: rsum = sum_m exp(score) ; waves split the 256 col-tiles ----
  float rsum[8];
  #pragma unroll
  for (int v = 0; v < 8; ++v) rsum[v] = 0.f;
  for (int mt = ch * 128; mt < ch * 128 + 128; ++mt) {
    v16h bm = *(const v16h*)(gbase + (long)mt * 512);
    v8f z = {};
    v8f s = WMMA_F16(a_s, bm, z);
    #pragma unroll
    for (int v = 0; v < 8; ++v) rsum[v] += __expf(s[v]);
  }
  #pragma unroll
  for (int off = 1; off < 16; off <<= 1) {
    #pragma unroll
    for (int v = 0; v < 8; ++v) rsum[v] += __shfl_xor(rsum[v], off, 32);
  }
  if (ln == 0) {
    #pragma unroll
    for (int v = 0; v < 8; ++v) ldsStats[rt][ch][v + 8 * lhi] = rsum[v];
  }
  __syncthreads();

  float rowinv[8];
  #pragma unroll
  for (int v = 0; v < 8; ++v)
    rowinv[v] = 1.f / (ldsStats[rt][0][v + 8 * lhi] + ldsStats[rt][1][v + 8 * lhi]);

  // ---- async-stage panel 0 (Hv 16KB + G 2KB), double buffered ----
  {
    const char* src = (const char*)hpan + tid * 64;          // panel jt=0
    unsigned dst = ldsHoff + (unsigned)tid * 80;
    #pragma unroll
    for (int i = 0; i < 4; ++i) async_copy_b128(dst + i * 16, src + i * 16);
    if (w < 4)                                               // 2KB, wave-uniform
      async_copy_b128(ldsGoff + (unsigned)tid * 16,
                      (const char*)(Gpk + (long)b * 131072) + tid * 16);
  }

  v8f acc[8];
  #pragma unroll
  for (int i = 0; i < 8; ++i) acc[i] = {};

  _Float16* myA = ldsA[w];
  float* attB = attention + (long)b * NN * NN;

  for (int jt = 0; jt < 128; ++jt) {
    const int buf = jt & 1;
    wait_async0();
    __syncthreads();
    if (jt + 1 < 128) {                       // issue next panel into buf^1
      const char* src = (const char*)(hpan + (long)(jt + 1) * 8192) + tid * 64;
      unsigned dst = ldsHoff + (unsigned)(buf ^ 1) * 20480u + (unsigned)tid * 80;
      #pragma unroll
      for (int i = 0; i < 4; ++i) async_copy_b128(dst + i * 16, src + i * 16);
      if (w < 4)
        async_copy_b128(ldsGoff + (unsigned)(buf ^ 1) * 2048u + (unsigned)tid * 16,
                        (const char*)(Gpk + (long)b * 131072 + (long)(jt + 1) * 1024) +
                            tid * 16);
    }

    // -- score recompute (2 tiles), attn vals, attention write, LDS pack --
    #pragma unroll
    for (int hf = 0; hf < 2; ++hf) {
      v16h bm = *(const v16h*)(&ldsG[buf][hf * 512 + lane * 16]);
      v8f z = {};
      v8f s = WMMA_F16(a_s, bm, z);
      float vals[8];
      #pragma unroll
      for (int v = 0; v < 8; ++v) vals[v] = __expf(s[v]) * rowinv[v];
      #pragma unroll
      for (int v = 0; v < 8; ++v)
        myA[(v + 8 * lhi) * 40 + hf * 16 + ln] = (_Float16)vals[v];
      if (ch == 0) {
        int m = (jt * 2 + hf) * 16 + ln;
        v4f* dst = (v4f*)(attB + (long)m * NN + n0 + 8 * lhi);
        v4f p0 = {vals[0], vals[1], vals[2], vals[3]};
        v4f p1 = {vals[4], vals[5], vals[6], vals[7]};
        __builtin_nontemporal_store(p0, dst);
        __builtin_nontemporal_store(p1, dst + 1);
      }
    }

    // -- attn tile as WMMA A operand (C-layout -> A-layout via LDS) --
    const _Float16* ap = myA + ln * 40 + 8 * lhi;
    v8h a0 = *(const v8h*)ap;
    v8h a1 = *(const v8h*)(ap + 16);
    v16h ao = __builtin_shufflevector(a0, a1, 0, 1, 2, 3, 4, 5, 6, 7,
                                      8, 9, 10, 11, 12, 13, 14, 15);

    // -- out^T += attn @ Hv^T : 8 C-tiles from the LDS panel --
    #pragma unroll
    for (int ct = 0; ct < 8; ++ct) {
      const _Float16* hb = &ldsH[buf][((ch * 8 + ct) * 16 + ln) * 40 + 8 * lhi];
      v8h b0 = *(const v8h*)hb;
      v8h b1 = *(const v8h*)(hb + 16);
      v16h bh = __builtin_shufflevector(b0, b1, 0, 1, 2, 3, 4, 5, 6, 7,
                                        8, 9, 10, 11, 12, 13, 14, 15);
      acc[ct] = WMMA_F16(ao, bh, acc[ct]);
    }
  }

  // ---- epilogue: out = gamma*acc + x (NT loads/stores) ----
  const float g = gamma[0];
  #pragma unroll
  for (int ct = 0; ct < 8; ++ct) {
    int c = ch * 128 + ct * 16 + ln;
    long base = ((long)b * CC + c) * NN + n0 + 8 * lhi;
    v4f x0 = __builtin_nontemporal_load((const v4f*)(x + base));
    v4f x1 = __builtin_nontemporal_load((const v4f*)(x + base) + 1);
    v8f a = acc[ct];
    v4f o0 = {g * a[0] + x0[0], g * a[1] + x0[1], g * a[2] + x0[2], g * a[3] + x0[3]};
    v4f o1 = {g * a[4] + x1[0], g * a[5] + x1[1], g * a[6] + x1[2], g * a[7] + x1[3]};
    __builtin_nontemporal_store(o0, (v4f*)(out + base));
    __builtin_nontemporal_store(o1, (v4f*)(out + base) + 1);
  }
}

// ---------------------------------------------------------------------------
extern "C" void kernel_launch(void* const* d_in, const int* in_sizes, int n_in,
                              void* d_out, int out_size, void* d_ws, size_t ws_size,
                              hipStream_t stream) {
  (void)in_sizes; (void)n_in; (void)out_size; (void)ws_size;
  const float* x     = (const float*)d_in[0];
  const float* Wf    = (const float*)d_in[1];
  const float* Wg    = (const float*)d_in[2];
  const float* Wh    = (const float*)d_in[3];
  const float* gamma = (const float*)d_in[4];

  float* out       = (float*)d_out;
  float* attention = out + (size_t)BB * CC * NN;   // outputs concatenated

  char* ws = (char*)d_ws;                          // needs ~20.2 MB
  _Float16* Apk = (_Float16*)(ws);                 //  2 MB
  _Float16* Gpk = (_Float16*)(ws + (2u << 20));    //  2 MB
  _Float16* Hcp = (_Float16*)(ws + (4u << 20));    // 16 MB
  _Float16* Wpk = (_Float16*)(ws + (20u << 20));   // 160 KB

  hipLaunchKernelGGL(pack_w_kernel, dim3(320), dim3(256), 0, stream, Wf, Wg, Wh, Wpk);
  hipLaunchKernelGGL(proj_kernel, dim3(BB * 256), dim3(256), 0, stream, x, Wpk, Apk, Gpk, Hcp);
  hipLaunchKernelGGL(attn_out_kernel, dim3(BB * 64), dim3(256), 0, stream,
                     Apk, Gpk, Hcp, x, gamma, out, attention);
}
